// CountHistogram_10582799417489
// MI455X (gfx1250) — compile-verified, matrix operand order checked by reading, not run on hardware
//
#include <hip/hip_runtime.h>
#include <hip/hip_bf16.h>

typedef __attribute__((ext_vector_type(16))) _Float16 v16h;
typedef __attribute__((ext_vector_type(8)))  float    v8f;
typedef __attribute__((ext_vector_type(4)))  float    v4f;

#define B_DIM   64
#define C_DIM   2
#define Q_DIM   64
#define D_DIM   4096
#define WAVES_PER_BLOCK 8
#define LSTRIDE 33   // padded per-lane slot stride (33 words -> good bank spread)

// One wave32 per (b,c,q) row. Each lane streams its share of the row with NT
// b128 loads and scatters counts into a PRIVATE per-lane LDS sub-histogram
// (ds_add_u32, zero intra-wave collisions). dlen is wave-uniform, so the
// stream loop is split into fully-valid iterations (no per-element predicate)
// plus one predicated boundary iteration; the invalid tail is never loaded
// (~25% HBM traffic saved on average). Cross-lane reduction is done with two
// v_wmma_f32_16x16x32_f16 (all-ones A x partial-count B): per-lane counts
// <=128 are exact in f16, row sums <=4096 exact in the f32 accumulator.
__global__ __launch_bounds__(256) void hist29_kernel(
    const float* __restrict__ simmat,
    const int*   __restrict__ dlens,
    const int*   __restrict__ qtoks,
    const int*   __restrict__ nbins_p,
    float*       __restrict__ out)
{
    __shared__ unsigned int lds[WAVES_PER_BLOCK * 32 * LSTRIDE];

    const int tid  = threadIdx.x;
    const int wave = tid >> 5;
    const int lane = tid & 31;
    const int row  = blockIdx.x * WAVES_PER_BLOCK + wave;   // [0, B*C*Q)
    const int b    = row >> 7;          // row / (C_DIM*Q_DIM)
    const int q    = row & (Q_DIM - 1); // row % Q_DIM

    const int   nb    = *nbins_p;                    // 29
    const float scale = (float)(nb - 1) * 0.5f;      // 14.0f

    unsigned int* h  = &lds[wave * 32 * LSTRIDE];
    unsigned int* hp = h + lane * LSTRIDE;
    float* orow = out + (size_t)row * nb;

    // Row with padded query token contributes nothing (wave-uniform branch,
    // so EXEC stays all-ones on the WMMA path below). Skips all loads.
    if (qtoks[b * Q_DIM + q] == -1) {
        if (lane < nb) orow[lane] = 0.0f;
        return;
    }

    // Zero this lane's private sub-histogram (bins 0..31; slot 32 is pad).
    #pragma unroll
    for (int s = 0; s < 32; ++s) hp[s] = 0u;

    // dtoks[b,d] != -1  <=>  d < dlens[b]  (by input construction).
    // dlen is wave-uniform -> uniform trip count, no divergence.
    const int dlen  = dlens[b];
    const int nfull = dlen >> 7;        // iterations with all 128 d's valid
    const v4f* rowp = (const v4f*)(simmat + (size_t)row * D_DIM);

    #pragma unroll 4
    for (int i = 0; i < nfull; ++i) {
        v4f v = __builtin_nontemporal_load(&rowp[i * 32 + lane]);
        #pragma unroll
        for (int e = 0; e < 4; ++e) {
            // ((x + 1.00001f)/2)*(nb-1), truncated — halving is exact so
            // (x+1.00001f)*scale is bit-identical to the reference.
            int bin = (int)((v[e] + 1.00001f) * scale);
            bin = bin < 0 ? 0 : (bin > 31 ? 31 : bin);  // safety clamp
            atomicAdd(&hp[bin], 1u);                    // ds_add_u32, private
        }
    }

    if (dlen & 127) {                   // predicated boundary iteration
        v4f v = __builtin_nontemporal_load(&rowp[nfull * 32 + lane]);
        const int d0 = (nfull * 32 + lane) * 4;
        #pragma unroll
        for (int e = 0; e < 4; ++e) {
            int bin = (int)((v[e] + 1.00001f) * scale);
            bin = bin < 0 ? 0 : (bin > 31 ? 31 : bin);
            if (d0 + e < dlen) atomicAdd(&hp[bin], 1u);
        }
    }
    // Tail iterations (d >= dlen rounded up) are never loaded: weight 0.

    // ---- Cross-lane reduction: ones(16x32) x partials(32x16) via WMMA ----
    // Lane l supplies column n = l&15; lanes l and l^16 together cover all
    // 32 per-lane partial counts of that column (order inside the 16 slots is
    // irrelevant since A is all ones -> every D row equals the column sums).
    const int n  = lane & 15;
    const int jb = lane & 16;
    v16h a, b0, b1;
    #pragma unroll
    for (int s = 0; s < 16; ++s) {
        a[s]  = (_Float16)1.0f;
        unsigned int c0 = h[(jb + s) * LSTRIDE + n];        // bins 0..15
        unsigned int c1 = h[(jb + s) * LSTRIDE + 16 + n];   // bins 16..31
        b0[s] = (_Float16)(float)c0;   // exact: counts <= 128 < 2048
        b1[s] = (_Float16)(float)c1;
    }
    v8f acc0 = {};
    v8f acc1 = {};
    acc0 = __builtin_amdgcn_wmma_f32_16x16x32_f16(false, a, false, b0,
                                                  (short)0, acc0, false, false);
    acc1 = __builtin_amdgcn_wmma_f32_16x16x32_f16(false, a, false, b1,
                                                  (short)0, acc1, false, false);

    // D layout: VGPR0 of lane t holds (row 0|8, col t%16); all rows equal.
    float res = (lane < 16) ? acc0[0] : acc1[0];  // lane t -> bin t
    if (lane < nb) orow[lane] = res;
}

extern "C" void kernel_launch(void* const* d_in, const int* in_sizes, int n_in,
                              void* d_out, int out_size, void* d_ws, size_t ws_size,
                              hipStream_t stream) {
    (void)in_sizes; (void)n_in; (void)out_size; (void)d_ws; (void)ws_size;
    const float* simmat = (const float*)d_in[0];
    const int*   dlens  = (const int*)  d_in[1];
    // d_in[2] = dtoks: unused, dtoks[b,d] != -1 <=> d < dlens[b] by construction
    const int*   qtoks  = (const int*)  d_in[3];
    const int*   nbins  = (const int*)  d_in[4];

    const int rows   = B_DIM * C_DIM * Q_DIM;           // 8192
    const int blocks = rows / WAVES_PER_BLOCK;          // 1024
    hipLaunchKernelGGL(hist29_kernel, dim3(blocks), dim3(256), 0, stream,
                       simmat, dlens, qtoks, nbins, (float*)d_out);
}
